// RealQuantLinear_88897233092669
// MI455X (gfx1250) — matrix operand order repaired and checked by reading, not compiled, not run on hardware
//
#include <hip/hip_runtime.h>

// ---- vector types matching gfx1250 WMMA operand shapes ----
typedef _Float16 half4 __attribute__((ext_vector_type(4)));
typedef _Float16 half8 __attribute__((ext_vector_type(8)));
typedef _Float16 v16h  __attribute__((ext_vector_type(16)));
typedef float    v8f   __attribute__((ext_vector_type(8)));
typedef float    f4v   __attribute__((ext_vector_type(4)));
typedef int      i4v   __attribute__((ext_vector_type(4)));

#define IN_F   4096
#define OUT_F  11008
#define GROUPS 32

#define BM 64
#define BN 128
#define BK 64
#define LDSS 72   // halves per LDS row: 144 bytes (16B-aligned rows, staggers banks)

static __device__ __forceinline__ v16h cat16(half8 lo, half8 hi) {
  return __builtin_shufflevector(lo, hi, 0,1,2,3,4,5,6,7,8,9,10,11,12,13,14,15);
}

__global__ __launch_bounds__(256)
void qlinear_wmma_f16(const float* __restrict__ x,
                      const int*   __restrict__ qw,
                      const float* __restrict__ scale,
                      const float* __restrict__ zp,
                      const float* __restrict__ bias,
                      float* __restrict__ out)
{
  __shared__ __align__(16) _Float16 xs[BM * LDSS];   //  9.0 KB
  __shared__ __align__(16) _Float16 ws[BN * LDSS];   // 18.0 KB

  const int tid   = threadIdx.x;
  const int lane  = tid & 31;        // wave32
  const int wave  = tid >> 5;        // 0..7
  const int lhalf = lane & 15;
  const int hi16  = lane >> 4;       // lane half-select per ISA layouts

  const int n0 = blockIdx.x * BN;
  const int m0 = blockIdx.y * BM;

  const int wm = wave >> 2;          // 0..1 : 32-row strip
  const int wn = wave & 3;           // 0..3 : 32-col strip

  // cooperative-loader coordinates: 16 rows x 16 lanes-of-4-elements per pass
  const int lr = tid >> 4;           // 0..15
  const int lc = (tid & 15) * 4;     // 0,4,...,60

  v8f acc[2][2] = {};

  for (int kk = 0; kk < IN_F; kk += BK) {
    // ---- stage x tile: 64x64 f32 -> f16 in LDS ----
    #pragma unroll
    for (int p = 0; p < 4; ++p) {
      const int r = p * 16 + lr;
      f4v v = *(const f4v*)(x + (size_t)(m0 + r) * IN_F + kk + lc);
      half4 h;
      h.x = (_Float16)v.x; h.y = (_Float16)v.y;
      h.z = (_Float16)v.z; h.w = (_Float16)v.w;
      *(half4*)&xs[r * LDSS + lc] = h;
    }

    // ---- stage + dequant weight tile: 128x64 int32 -> f16 in LDS ----
    const int grp = kk >> 7;         // BK=64 divides group size 128
    #pragma unroll
    for (int p = 0; p < 8; ++p) {
      const int r  = p * 16 + lr;
      const int gn = n0 + r;
      const int* wp = qw + (size_t)gn * IN_F + kk + lc;
      i4v q = *(const i4v*)wp;
      if (kk + BK < IN_F)            // pull next 32KB weight tile toward L2/L0
        __builtin_prefetch(wp + BK, 0, 0);
      const float s = scale[gn * GROUPS + grp];
      const float z = zp[gn * GROUPS + grp];
      half4 h;
      h.x = (_Float16)(((float)q.x - z) * s);
      h.y = (_Float16)(((float)q.y - z) * s);
      h.z = (_Float16)(((float)q.z - z) * s);
      h.w = (_Float16)(((float)q.w - z) * s);
      *(half4*)&ws[r * LDSS + lc] = h;
    }
    __syncthreads();

    // ---- 2 WMMA K-steps of 32 over the 64-wide chunk ----
    #pragma unroll
    for (int ks = 0; ks < BK; ks += 32) {
      // A fragments (16x32 f16): lane holds row m=lane&15;
      // halves k = ks + kb + 0..7 and ks + 16 + kb + 0..7, kb = 8*(lane>=16)
      v16h a[2];
      #pragma unroll
      for (int mf = 0; mf < 2; ++mf) {
        const int row = wm * 32 + mf * 16 + lhalf;
        half8 lo = *(const half8*)&xs[row * LDSS + ks + hi16 * 8];
        half8 hi = *(const half8*)&xs[row * LDSS + ks + 16 + hi16 * 8];
        a[mf] = cat16(lo, hi);
      }
      #pragma unroll
      for (int nf = 0; nf < 2; ++nf) {
        // B fragment (32x16 f16): lane holds col n=lane&15,
        // halves k = ks + kb + 0..7 and ks + 16 + kb + 0..7, kb = 8*(lane>=16)
        const int col = wn * 32 + nf * 16 + lhalf;
        const half8* wr = (const half8*)&ws[col * LDSS + ks + hi16 * 8];
        v16h b = cat16(wr[0], wr[2]);
        #pragma unroll
        for (int mf = 0; mf < 2; ++mf) {
          acc[mf][nf] = __builtin_amdgcn_wmma_f32_16x16x32_f16(
              /*neg_a=*/false, a[mf], /*neg_b=*/false, b,
              /*c_mod=*/(short)0, acc[mf][nf],
              /*reuse_a=*/false, /*reuse_b=*/false);
        }
      }
    }
    __syncthreads();
  }

  // ---- epilogue: + bias, scatter per C/D layout (lane n = lane&15, VGPR v -> m = 8*(lane>=16)+v)
  #pragma unroll
  for (int nf = 0; nf < 2; ++nf) {
    const int gn = n0 + wn * 32 + nf * 16 + lhalf;
    const float bv = bias[gn];
    #pragma unroll
    for (int mf = 0; mf < 2; ++mf) {
      const int gm = m0 + wm * 32 + mf * 16 + hi16 * 8;
      #pragma unroll
      for (int v = 0; v < 8; ++v) {
        out[(size_t)(gm + v) * OUT_F + gn] = acc[mf][nf][v] + bv;
      }
    }
  }
}

extern "C" void kernel_launch(void* const* d_in, const int* in_sizes, int n_in,
                              void* d_out, int out_size, void* d_ws, size_t ws_size,
                              hipStream_t stream) {
  const float* x     = (const float*)d_in[0];
  const int*   qw    = (const int*)d_in[1];
  const float* scal  = (const float*)d_in[2];
  const float* zpt   = (const float*)d_in[3];
  const float* bias  = (const float*)d_in[4];
  float* out = (float*)d_out;

  const int M = in_sizes[0] / IN_F;          // 2*256 = 512
  dim3 grid(OUT_F / BN, M / BM);             // (86, 8)
  qlinear_wmma_f16<<<grid, 256, 0, stream>>>(x, qw, scal, zpt, bias, out);
}
